// DocREModel_82025285419178
// MI455X (gfx1250) — matrix operand order repaired and checked by reading, compiled, not run on hardware
//
#include <hip/hip_runtime.h>
#include <stdint.h>

#define B_    2
#define L_    1024
#define H_    768
#define E_    32
#define M_    4
#define EMB_  768
#define BLK_  64
#define G_    12
#define NL_   97
#define PAIRS_ 2048
#define KTOT_   (EMB_*BLK_)   /* 49152 */
#define KB_TOT_ (KTOT_/32)    /* 1536  */
#define NT_TOT_ (EMB_/16)     /* 48    */

typedef __attribute__((ext_vector_type(16))) __bf16    v16bf;
typedef __attribute__((ext_vector_type(8)))  float     v8f;
typedef __attribute__((ext_vector_type(8)))  uint32_t  v8u;

static __device__ __forceinline__ uint16_t f2bf(float f) {
  uint32_t u = __builtin_bit_cast(uint32_t, f);
  u += 0x7FFFu + ((u >> 16) & 1u);   // round-to-nearest-even
  return (uint16_t)(u >> 16);
}

// ---------------------------------------------------------------------------
// Repack W_ext (49152 x 768 f32, row major) into bf16 WMMA-B fragments.
// Fragment (kb, nt): 32 lanes x 8 dwords. Lane l: n = nt*16 + (l&15), hi = l>>4.
// Dword v packs B[kb*32 + hi*16 + 2v, n] (lo16) and B[... + 2v + 1, n] (hi16).
// ---------------------------------------------------------------------------
__global__ __launch_bounds__(256) void k_convert_wext(const float* __restrict__ W,
                                                      uint32_t* __restrict__ Wb) {
  size_t idx = (size_t)blockIdx.x * 256 + threadIdx.x;
  const size_t total = (size_t)KB_TOT_ * NT_TOT_ * 32 * 8;
  if (idx >= total) return;
  int v = (int)(idx & 7);  size_t r = idx >> 3;
  int lane = (int)(r & 31); r >>= 5;
  int nt = (int)(r % NT_TOT_);
  size_t kb = r / NT_TOT_;
  int n  = nt * 16 + (lane & 15);
  int hi = lane >> 4;
  size_t k = kb * 32 + (size_t)(hi * 16 + 2 * v);
  uint16_t lo = f2bf(W[k * EMB_ + n]);
  uint16_t hh = f2bf(W[(k + 1) * EMB_ + n]);
  Wb[idx] = (uint32_t)lo | ((uint32_t)hh << 16);
}

// ---------------------------------------------------------------------------
// Entity pooling: ent[b,e,:] = mean over mentions of 0.5*(seq[s+1]+seq[e]);
// also extracts cls[b,:] = seq[b,0,:].
// ---------------------------------------------------------------------------
__global__ __launch_bounds__(256) void k_entities(const float* __restrict__ seq,
    const int* __restrict__ starts, const int* __restrict__ ends,
    const float* __restrict__ mmask, float* __restrict__ ent, float* __restrict__ cls) {
  int be = blockIdx.x;           // 0..63
  int b  = be >> 5;
  int t  = threadIdx.x;
  float a0 = 0.f, a1 = 0.f, a2 = 0.f, wsum = 0.f;
  for (int mm = 0; mm < M_; ++mm) {
    int idx = be * M_ + mm;
    int s = starts[idx] + 1;
    int e = ends[idx];
    float w = mmask[idx];
    wsum += w;
    const float* r1 = seq + ((size_t)b * L_ + s) * H_;
    const float* r2 = seq + ((size_t)b * L_ + e) * H_;
    a0 += w * 0.5f * (r1[t]       + r2[t]);
    a1 += w * 0.5f * (r1[t + 256] + r2[t + 256]);
    a2 += w * 0.5f * (r1[t + 512] + r2[t + 512]);
  }
  float inv = 1.f / fmaxf(wsum, 1.f);
  ent[(size_t)be * H_ + t]       = a0 * inv;
  ent[(size_t)be * H_ + t + 256] = a1 * inv;
  ent[(size_t)be * H_ + t + 512] = a2 * inv;
  if ((be & 31) == 0) {
    const float* c0 = seq + (size_t)b * L_ * H_;
    cls[(size_t)b * H_ + t]       = c0[t];
    cls[(size_t)b * H_ + t + 256] = c0[t + 256];
    cls[(size_t)b * H_ + t + 512] = c0[t + 512];
  }
}

// ---------------------------------------------------------------------------
// Six small projections: P1/P2 = ent @ W_head[0:768]/[768:1536];
// Q1/Q2 = ent @ W_tail slices; P3/Q3 = cls @ third slices.
// ---------------------------------------------------------------------------
__global__ __launch_bounds__(256) void k_proj(const float* __restrict__ ent,
    const float* __restrict__ cls, const float* __restrict__ Wh, const float* __restrict__ Wt,
    float* __restrict__ P1, float* __restrict__ P2, float* __restrict__ Q1,
    float* __restrict__ Q2, float* __restrict__ P3, float* __restrict__ Q3) {
  __shared__ float xrow[EMB_];
  int bid = blockIdx.x;          // 0..259
  const float* x; const float* W; float* o;
  if (bid < 64)       { int r = bid;       x = ent + (size_t)r*EMB_; W = Wh;                         o = P1 + (size_t)r*EMB_; }
  else if (bid < 128) { int r = bid - 64;  x = ent + (size_t)r*EMB_; W = Wh + (size_t)EMB_*EMB_;     o = P2 + (size_t)r*EMB_; }
  else if (bid < 192) { int r = bid - 128; x = ent + (size_t)r*EMB_; W = Wt;                         o = Q1 + (size_t)r*EMB_; }
  else if (bid < 256) { int r = bid - 192; x = ent + (size_t)r*EMB_; W = Wt + (size_t)EMB_*EMB_;     o = Q2 + (size_t)r*EMB_; }
  else if (bid < 258) { int r = bid - 256; x = cls + (size_t)r*EMB_; W = Wh + (size_t)2*EMB_*EMB_;   o = P3 + (size_t)r*EMB_; }
  else                { int r = bid - 258; x = cls + (size_t)r*EMB_; W = Wt + (size_t)2*EMB_*EMB_;   o = Q3 + (size_t)r*EMB_; }
  for (int k = threadIdx.x; k < EMB_; k += 256) xrow[k] = x[k];
  __syncthreads();
  for (int c = threadIdx.x; c < EMB_; c += 256) {
    float a = 0.f;
    for (int k = 0; k < EMB_; ++k) a = fmaf(xrow[k], W[(size_t)k * EMB_ + c], a);
    o[c] = a;
  }
}

// ---------------------------------------------------------------------------
// hs[p] = tanh(P1[x] + P2[y] + P3[b] + b_head) ; ts[p] = tanh(Q1[y]+Q2[x]+Q3[b]+b_tail)
// stored bf16 for WMMA consumption.
// ---------------------------------------------------------------------------
__global__ __launch_bounds__(256) void k_pairvec(const float* __restrict__ P1,
    const float* __restrict__ P2, const float* __restrict__ P3, const float* __restrict__ Q1,
    const float* __restrict__ Q2, const float* __restrict__ Q3, const float* __restrict__ bh,
    const float* __restrict__ bt, uint16_t* __restrict__ hsb, uint16_t* __restrict__ tsb) {
  int p = blockIdx.x;            // 0..2047
  int b = p >> 10, x = (p >> 5) & 31, y = p & 31;
  const float* p1 = P1 + (size_t)(b * 32 + x) * EMB_;
  const float* p2 = P2 + (size_t)(b * 32 + y) * EMB_;
  const float* p3 = P3 + (size_t)b * EMB_;
  const float* q1 = Q1 + (size_t)(b * 32 + y) * EMB_;
  const float* q2 = Q2 + (size_t)(b * 32 + x) * EMB_;
  const float* q3 = Q3 + (size_t)b * EMB_;
  uint16_t* ho = hsb + (size_t)p * EMB_;
  uint16_t* to = tsb + (size_t)p * EMB_;
  for (int c = threadIdx.x; c < EMB_; c += 256) {
    ho[c] = f2bf(tanhf(p1[c] + p2[c] + p3[c] + bh[c]));
    to[c] = f2bf(tanhf(q1[c] + q2[c] + q3[c] + bt[c]));
  }
}

// ---------------------------------------------------------------------------
// Main fused bilinear GEMM: feat[p,n] = relu( sum_{g,i,j} hs[p,gi]*ts[p,gj]*Wext[(gi)*64+j, n] + b_ext )
// then LayerNorm and logits = ln @ W_cls, per 16-pair tile.
// M=2048 (16/block), K=49152, N=768 (96 cols/wave as 6 WMMA N-tiles).
// ---------------------------------------------------------------------------
__global__ __launch_bounds__(256) void k_pair_gemm(
    const uint16_t* __restrict__ hsb, const uint16_t* __restrict__ tsb,
    const __bf16* __restrict__ Wb, const float* __restrict__ b_ext,
    const float* __restrict__ ln_g, const float* __restrict__ ln_b,
    const float* __restrict__ W_cls, float* __restrict__ out) {
  __shared__ __bf16 hs_lds[16 * EMB_];
  __shared__ __bf16 ts_lds[16 * EMB_];
  __shared__ float  feat_lds[16 * EMB_];
  __shared__ float  red[512];
  __shared__ float  stats[32];

  const int tid  = threadIdx.x;
  const int wave = tid >> 5;
  const int lane = tid & 31;
  const int mt   = blockIdx.x;   // 0..127 (16 pairs each)

  { // stage hs/ts tiles into LDS
    const uint32_t* hg = (const uint32_t*)(hsb + (size_t)mt * 16 * EMB_);
    const uint32_t* tg = (const uint32_t*)(tsb + (size_t)mt * 16 * EMB_);
    uint32_t* hl = (uint32_t*)hs_lds;
    uint32_t* tl = (uint32_t*)ts_lds;
    for (int i = tid; i < 16 * EMB_ / 2; i += 256) { hl[i] = hg[i]; tl[i] = tg[i]; }
  }
  __syncthreads();

  const int m   = lane & 15;
  const int hi  = lane >> 4;
  const int nt0 = wave * 6;      // 8 waves * 6 tiles = 48 N-tiles = 768 cols

  v8f acc[6];
  #pragma unroll
  for (int t = 0; t < 6; ++t) acc[t] = (v8f){0.f,0.f,0.f,0.f,0.f,0.f,0.f,0.f};

  const __bf16* hsrow = hs_lds + m * EMB_;
  const __bf16* tsrow = ts_lds + m * EMB_;

  size_t wboff = (size_t)nt0 * 512 + (size_t)lane * 16;   // elements
  const size_t wbstep = (size_t)NT_TOT_ * 512;

  #pragma unroll 1
  for (int g = 0; g < G_; ++g) {
    #pragma unroll 1
    for (int i = 0; i < BLK_; ++i) {
      uint32_t hv   = *(const uint16_t*)(hsrow + g * BLK_ + i);
      uint32_t hrep = hv * 0x10001u;                      // bf16 scalar in both halves
      #pragma unroll
      for (int j0 = 0; j0 < BLK_; j0 += 32) {
        // A fragment: contiguous K order; lanes 0-15 K{0..15}, lanes 16-31 K{16..31}
        uint4 t0 = *(const uint4*)(tsrow + (g * BLK_ + j0 + hi * 16));
        uint4 t1 = *(const uint4*)(tsrow + (g * BLK_ + j0 + hi * 16 + 8));
        uint32_t a0, a1, a2, a3, a4, a5, a6, a7;
        asm("v_pk_mul_bf16 %0, %1, %2" : "=v"(a0) : "v"(hrep), "v"(t0.x));
        asm("v_pk_mul_bf16 %0, %1, %2" : "=v"(a1) : "v"(hrep), "v"(t0.y));
        asm("v_pk_mul_bf16 %0, %1, %2" : "=v"(a2) : "v"(hrep), "v"(t0.z));
        asm("v_pk_mul_bf16 %0, %1, %2" : "=v"(a3) : "v"(hrep), "v"(t0.w));
        asm("v_pk_mul_bf16 %0, %1, %2" : "=v"(a4) : "v"(hrep), "v"(t1.x));
        asm("v_pk_mul_bf16 %0, %1, %2" : "=v"(a5) : "v"(hrep), "v"(t1.y));
        asm("v_pk_mul_bf16 %0, %1, %2" : "=v"(a6) : "v"(hrep), "v"(t1.z));
        asm("v_pk_mul_bf16 %0, %1, %2" : "=v"(a7) : "v"(hrep), "v"(t1.w));
        v8u au = {a0, a1, a2, a3, a4, a5, a6, a7};
        v16bf afrag = __builtin_bit_cast(v16bf, au);
        const __bf16* wp = Wb + wboff;
        #pragma unroll
        for (int t = 0; t < 6; ++t) {
          v16bf bfrag = *(const v16bf*)(wp + (size_t)t * 512);
          acc[t] = __builtin_amdgcn_wmma_f32_16x16x32_bf16(
              false, afrag, false, bfrag, (short)0, acc[t], false, false);
        }
        wboff += wbstep;
      }
    }
  }

  // Epilogue 1: bias + ReLU -> feat_lds. C layout: VGPR r holds M=r (lanes 0-15) / M=r+8 (16-31).
  #pragma unroll
  for (int t = 0; t < 6; ++t) {
    int n = (nt0 + t) * 16 + m;
    float bias = b_ext[n];
    #pragma unroll
    for (int r = 0; r < 8; ++r) {
      int mm = r + hi * 8;
      feat_lds[mm * EMB_ + n] = fmaxf(acc[t][r] + bias, 0.f);
    }
  }
  __syncthreads();

  // Epilogue 2: LayerNorm stats (16 threads per pair)
  {
    int p = tid >> 4, sub = tid & 15;
    float s = 0.f, ss = 0.f;
    for (int n = sub; n < EMB_; n += 16) {
      float v = feat_lds[p * EMB_ + n];
      s += v; ss += v * v;
    }
    red[tid] = s; red[256 + tid] = ss;
    __syncthreads();
    if (sub == 0) {
      float S = 0.f, SS = 0.f;
      for (int q = 0; q < 16; ++q) { S += red[(p << 4) + q]; SS += red[256 + (p << 4) + q]; }
      float mu  = S * (1.f / EMB_);
      float var = SS * (1.f / EMB_) - mu * mu;
      stats[p * 2]     = mu;
      stats[p * 2 + 1] = rsqrtf(var + 1e-12f);
    }
    __syncthreads();
  }

  // Epilogue 3: normalize in place
  for (int idx = tid; idx < 16 * EMB_; idx += 256) {
    int pp = idx / EMB_, n = idx - pp * EMB_;
    feat_lds[idx] = (feat_lds[idx] - stats[pp * 2]) * stats[pp * 2 + 1] * ln_g[n] + ln_b[n];
  }
  __syncthreads();

  // Epilogue 4: logits = ln @ W_cls (768 -> 97)
  for (int o = tid; o < 16 * NL_; o += 256) {
    int pp = o / NL_, nl = o - pp * NL_;
    const float* row = feat_lds + pp * EMB_;
    float a = 0.f;
    for (int k = 0; k < EMB_; ++k) a = fmaf(row[k], W_cls[(size_t)k * NL_ + nl], a);
    out[((size_t)mt * 16 + pp) * NL_ + nl] = a;
  }
}

// ---------------------------------------------------------------------------
extern "C" void kernel_launch(void* const* d_in, const int* in_sizes, int n_in,
                              void* d_out, int out_size, void* d_ws, size_t ws_size,
                              hipStream_t stream) {
  const float* seq    = (const float*)d_in[0];
  const int*   starts = (const int*)d_in[1];
  const int*   ends   = (const int*)d_in[2];
  const float* mmask  = (const float*)d_in[3];
  const float* W_head = (const float*)d_in[4];
  const float* b_head = (const float*)d_in[5];
  const float* W_tail = (const float*)d_in[6];
  const float* b_tail = (const float*)d_in[7];
  const float* W_ext  = (const float*)d_in[8];
  const float* b_ext  = (const float*)d_in[9];
  const float* ln_g   = (const float*)d_in[10];
  const float* ln_b   = (const float*)d_in[11];
  const float* W_cls  = (const float*)d_in[12];
  (void)in_sizes; (void)n_in; (void)out_size; (void)ws_size;

  char* ws = (char*)d_ws;
  size_t off = 0;
  auto alloc = [&](size_t bytes) -> void* {
    void* p = ws + off;
    off += (bytes + 255) & ~(size_t)255;
    return p;
  };

  uint16_t* Wb  = (uint16_t*)alloc((size_t)KTOT_ * EMB_ * 2);   // 75.5 MB bf16 fragments
  uint16_t* hsb = (uint16_t*)alloc((size_t)PAIRS_ * EMB_ * 2);
  uint16_t* tsb = (uint16_t*)alloc((size_t)PAIRS_ * EMB_ * 2);
  float* ent = (float*)alloc((size_t)B_ * E_ * EMB_ * 4);
  float* cls = (float*)alloc((size_t)B_ * EMB_ * 4);
  float* P1  = (float*)alloc((size_t)B_ * E_ * EMB_ * 4);
  float* P2  = (float*)alloc((size_t)B_ * E_ * EMB_ * 4);
  float* Q1  = (float*)alloc((size_t)B_ * E_ * EMB_ * 4);
  float* Q2  = (float*)alloc((size_t)B_ * E_ * EMB_ * 4);
  float* P3  = (float*)alloc((size_t)B_ * EMB_ * 4);
  float* Q3  = (float*)alloc((size_t)B_ * EMB_ * 4);

  const size_t cvt_total = (size_t)KB_TOT_ * NT_TOT_ * 32 * 8;
  k_convert_wext<<<(unsigned)((cvt_total + 255) / 256), 256, 0, stream>>>(W_ext, (uint32_t*)Wb);
  k_entities<<<B_ * E_, 256, 0, stream>>>(seq, starts, ends, mmask, ent, cls);
  k_proj<<<260, 256, 0, stream>>>(ent, cls, W_head, W_tail, P1, P2, Q1, Q2, P3, Q3);
  k_pairvec<<<PAIRS_, 256, 0, stream>>>(P1, P2, P3, Q1, Q2, Q3, b_head, b_tail, hsb, tsb);
  k_pair_gemm<<<PAIRS_ / 16, 256, 0, stream>>>(hsb, tsb, (const __bf16*)Wb,
                                               b_ext, ln_g, ln_b, W_cls, (float*)d_out);
}
